// AttentionSimilarity_64948495450527
// MI455X (gfx1250) — matrix operand order repaired and checked by reading, not compiled, not run on hardware
//
#include <hip/hip_runtime.h>

typedef __attribute__((ext_vector_type(16))) __bf16        v16bf;
typedef __attribute__((ext_vector_type(8)))  __bf16        v8bf;
typedef __attribute__((ext_vector_type(4)))  __bf16        v4bf;
typedef __attribute__((ext_vector_type(8)))  float         v8f;
typedef __attribute__((ext_vector_type(4)))  unsigned int  u32x4;
typedef __attribute__((ext_vector_type(8)))  unsigned int  u32x8;
typedef __attribute__((ext_vector_type(2)))  unsigned int  u32x2;
typedef __attribute__((ext_vector_type(4)))  float         f32x4;

#define TOKENS 4096   // B*S
#define DMODEL 768
#define SEQ    2048
#define NHEADS 12
#define HDIM   64

// ---------------------------------------------------------------------------
// Tensor Data Mover helpers (CDNA5 §8 D# descriptor, 2D tiles, bf16 payload)
// ---------------------------------------------------------------------------
__device__ __forceinline__ u32x4 tdm_group0(unsigned lds_addr, const void* gaddr) {
  unsigned long long ga = (unsigned long long)(size_t)gaddr;
  u32x4 g;
  g[0] = 1u;                                        // count=1 (valid), user mode
  g[1] = lds_addr;                                  // LDS byte address
  g[2] = (unsigned)(ga & 0xFFFFFFFFu);              // global_addr[31:0]
  g[3] = (unsigned)((ga >> 32) & 0x01FFFFFFu)       // global_addr[56:32]
       | (2u << 30);                                // type=2 ("image")
  return g;
}

// data_size=1 (2 bytes/elem); dims & strides in elements; 2D tile (tile_dim2=0)
__device__ __forceinline__ u32x8 tdm_group1_2d(unsigned tensor_d0, unsigned tensor_d1,
                                               unsigned tile_d0,   unsigned tile_d1,
                                               unsigned stride_d0) {
  u32x8 g;
  g[0] = (1u << 16);                                        // data_size=2B
  g[1] = (tensor_d0 & 0xFFFFu) << 16;                       // tensor_dim0[15:0]
  g[2] = ((tensor_d0 >> 16) & 0xFFFFu)
       | ((tensor_d1 & 0xFFFFu) << 16);                     // dim0 hi | dim1 lo
  g[3] = ((tensor_d1 >> 16) & 0xFFFFu)
       | ((tile_d0 & 0xFFFFu) << 16);                       // dim1 hi | tile_dim0
  g[4] = (tile_d1 & 0xFFFFu);                               // tile_dim1, tile_dim2=0
  g[5] = stride_d0;                                         // tensor_dim0_stride lo32
  g[6] = 0;                                                 // stride hi, dim1_stride lo
  g[7] = 0;
  return g;
}

__device__ __forceinline__ void tdm_load_2d(u32x4 g0, u32x8 g1) {
  asm volatile("tensor_load_to_lds %0, %1" :: "s"(g0), "s"(g1) : "memory");
}

// ---------------------------------------------------------------------------
// WMMA fragment helpers (16-bit, contiguous k order in A and B)
// ---------------------------------------------------------------------------
union FragU { u32x4 q[2]; v16bf v; };

// A 16x32 (MxK): lanes 0-15 row=lane, K 0..15; lanes 16-31 K 16..31 (contiguous)
__device__ __forceinline__ v16bf load_a_frag(const __bf16* base, int lane,
                                             int row_stride, int dofs) {
  const int row = lane & 15;
  const int kb  = (lane < 16) ? 0 : 16;
  const __bf16* p = base + row * row_stride + dofs + kb;
  FragU f;
  f.q[0] = *(const u32x4*)(p);
  f.q[1] = *(const u32x4*)(p + 8);
  return f.v;
}

// B 32x16 (KxN): lane holds column N=lane&15; K = (lane<16?0:16)+e, contiguous
__device__ __forceinline__ v16bf load_b_frag(const __bf16* p) {
  FragU f;
  f.q[0] = *(const u32x4*)(p);
  f.q[1] = *(const u32x4*)(p + 8);
  return f.v;
}

__device__ __forceinline__ v8f wmma_bf16(v16bf a, v16bf b, v8f c) {
  return __builtin_amdgcn_wmma_f32_16x16x32_bf16(false, a, false, b,
                                                 (short)0, c, false, false);
}

// ---------------------------------------------------------------------------
// Kernel 0: f32 -> bf16 bulk convert (native v_cvt path)
// ---------------------------------------------------------------------------
__global__ __launch_bounds__(256) void cvt_bf16_kernel(const float* __restrict__ in,
                                                       __bf16* __restrict__ outp,
                                                       int n) {
  const int i = (blockIdx.x * 256 + threadIdx.x) * 8;
  if (i >= n) return;
  f32x4 a = *(const f32x4*)(in + i);
  f32x4 b = *(const f32x4*)(in + i + 4);
  v8bf o;
  v4bf lo = __builtin_convertvector(a, v4bf);
  v4bf hi = __builtin_convertvector(b, v4bf);
  o[0]=lo[0]; o[1]=lo[1]; o[2]=lo[2]; o[3]=lo[3];
  o[4]=hi[0]; o[5]=hi[1]; o[6]=hi[2]; o[7]=hi[3];
  *(v8bf*)(outp + i) = o;
}

// ---------------------------------------------------------------------------
// Kernel 1: Q = x*Wq^T + bq, K = x*Wk^T + bk (bf16 in, bf16 out, f32 acc).
// grid = (TOKENS/16, 12): y 0..5 -> Q col-groups of 128, y 6..11 -> K.
// Tiles staged to LDS by the Tensor Data Mover (wave 0 issues, all waves sync).
// ---------------------------------------------------------------------------
__global__ __launch_bounds__(256) void qk_proj_kernel(
    const __bf16* __restrict__ xb,
    const __bf16* __restrict__ wqb, const float* __restrict__ bq,
    const __bf16* __restrict__ wkb, const float* __restrict__ bk,
    __bf16* __restrict__ outQ, __bf16* __restrict__ outK)
{
  __shared__ alignas(16) __bf16 lx[16 * 32];    // x tile [16 rows][32 d]
  __shared__ alignas(16) __bf16 lw[128 * 32];   // W tile [128 cols][32 d]

  const int t    = threadIdx.x;
  const int wave = t >> 5;
  const int lane = t & 31;

  const int  rowb = blockIdx.x * 16;
  const int  cg   = blockIdx.y;
  const bool isK  = (cg >= 6);
  const int  colb = (isK ? (cg - 6) : cg) * 128;
  const __bf16* __restrict__ W    = isK ? wkb : wqb;
  const float*  __restrict__ bias = isK ? bk : bq;
  __bf16* __restrict__ out = isK ? outK : outQ;

  const unsigned lds_lx = (unsigned)(size_t)&lx[0];
  const unsigned lds_lw = (unsigned)(size_t)&lw[0];
  const u32x8 g1x = tdm_group1_2d(DMODEL, TOKENS, 32, 16,  DMODEL);
  const u32x8 g1w = tdm_group1_2d(DMODEL, DMODEL, 32, 128, DMODEL);

  v8f acc;
  #pragma unroll
  for (int i = 0; i < 8; ++i) acc[i] = 0.0f;

  for (int kc = 0; kc < DMODEL; kc += 32) {
    __syncthreads();   // previous iteration's fragment reads complete
    if (wave == 0) {
      tdm_load_2d(tdm_group0(lds_lx, xb + (size_t)rowb * DMODEL + kc), g1x);
      tdm_load_2d(tdm_group0(lds_lw, W  + (size_t)colb * DMODEL + kc), g1w);
      __builtin_amdgcn_s_wait_tensorcnt(0);
    }
    __syncthreads();

    v16bf a = load_a_frag(lx, lane, 32, 0);
    v16bf b = load_b_frag(&lw[(wave * 16 + (lane & 15)) * 32 + ((lane < 16) ? 0 : 16)]);
    acc = wmma_bf16(a, b, acc);
  }

  // epilogue: +bias, f32 -> bf16 (native cvt), store
  const int col = colb + wave * 16 + (lane & 15);
  const float bb = bias[col];
  const int rb = (lane < 16) ? 0 : 8;
  #pragma unroll
  for (int r = 0; r < 8; ++r)
    out[(size_t)(rowb + rb + r) * DMODEL + col] = (__bf16)(acc[r] + bb);
}

// ---------------------------------------------------------------------------
// Kernel 2: per (b, 16-row q tile): loop heads, WMMA scores -> 128KB LDS buffer,
// softmax over k, accumulate mean over heads into out (block-exclusive rows).
// grid = (SEQ/16, 2); 8 waves, wave w owns k cols [w*256, w*256+256).
// ---------------------------------------------------------------------------
__global__ __launch_bounds__(256) void attn_kernel(
    const __bf16* __restrict__ Q,
    const __bf16* __restrict__ K,
    float* __restrict__ out)
{
  __shared__ alignas(16) float  sbuf[16 * SEQ];   // 128 KB score buffer
  __shared__ alignas(16) __bf16 qt[16 * HDIM];    // Q tile
  __shared__ float red[16 * 16];
  __shared__ float rowstat[16];

  const int t    = threadIdx.x;
  const int wave = t >> 5;
  const int lane = t & 31;
  const int b    = blockIdx.y;
  const int qb   = blockIdx.x;

  const size_t qrow0 = (size_t)b * SEQ + (size_t)qb * 16;
  float* __restrict__ outrow = out + qrow0 * SEQ;

  const float scale  = 0.125f;          // 1/sqrt(HDIM)
  const float hscale = 1.0f / (float)NHEADS;

  const unsigned lds_qt = (unsigned)(size_t)&qt[0];
  const u32x8 g1q = tdm_group1_2d(DMODEL, TOKENS, HDIM, 16, DMODEL);

  const int row = t >> 4;   // softmax: 16 threads per row
  const int seg = t & 15;

  for (int h = 0; h < NHEADS; ++h) {
    // stage Q tile (16x64 bf16) via Tensor Data Mover
    if (wave == 0) {
      tdm_load_2d(tdm_group0(lds_qt, Q + qrow0 * DMODEL + h * HDIM), g1q);
      __builtin_amdgcn_s_wait_tensorcnt(0);
    }
    __syncthreads();

    // phase 1: raw scores for this wave's 256 k-columns
    const int kw0 = wave * 256;
    #pragma unroll 1
    for (int nt = 0; nt < 16; ++nt) {
      const int kcol = kw0 + nt * 16;
      const __bf16* kp = K + ((size_t)b * SEQ + kcol + (lane & 15)) * DMODEL
                           + h * HDIM + ((lane < 16) ? 0 : 16);
      __builtin_prefetch(kp + (size_t)16 * DMODEL, 0, 1);   // next k tile

      v8f c;
      #pragma unroll
      for (int i = 0; i < 8; ++i) c[i] = 0.0f;

      v16bf a0 = load_a_frag(qt, lane, HDIM, 0);
      v16bf b0 = load_b_frag(kp);
      c = wmma_bf16(a0, b0, c);
      v16bf a1 = load_a_frag(qt, lane, HDIM, 32);
      v16bf b1 = load_b_frag(kp + 32);
      c = wmma_bf16(a1, b1, c);

      const int n  = lane & 15;
      const int rb = (lane < 16) ? 0 : 8;
      #pragma unroll
      for (int r = 0; r < 8; ++r)
        sbuf[(rb + r) * SEQ + kcol + n] = c[r] * scale;
    }
    __syncthreads();

    // phase 2: row-wise softmax over 2048 cols
    float m = -3.0e38f;
    for (int j = seg; j < SEQ; j += 16)
      m = fmaxf(m, sbuf[row * SEQ + j]);
    red[row * 16 + seg] = m;
    __syncthreads();
    if (seg == 0) {
      float mm = red[row * 16];
      #pragma unroll
      for (int i = 1; i < 16; ++i) mm = fmaxf(mm, red[row * 16 + i]);
      rowstat[row] = mm;
    }
    __syncthreads();
    const float rmax = rowstat[row];
    float s = 0.0f;
    for (int j = seg; j < SEQ; j += 16) {
      float e = __expf(sbuf[row * SEQ + j] - rmax);
      sbuf[row * SEQ + j] = e;
      s += e;
    }
    red[row * 16 + seg] = s;
    __syncthreads();
    if (seg == 0) {
      float ss = 0.0f;
      #pragma unroll
      for (int i = 0; i < 16; ++i) ss += red[row * 16 + i];
      rowstat[row] = ss;
    }
    __syncthreads();

    // accumulate head mean into out (rows owned exclusively by this block)
    const float inv = hscale / rowstat[row];
    for (int j = seg; j < SEQ; j += 16) {
      float v = sbuf[row * SEQ + j] * inv;
      float* po = outrow + (size_t)row * SEQ + j;
      if (h == 0) *po = v;
      else        *po += v;
    }
    __syncthreads();
  }
}

// ---------------------------------------------------------------------------
extern "C" void kernel_launch(void* const* d_in, const int* in_sizes, int n_in,
                              void* d_out, int out_size, void* d_ws, size_t ws_size,
                              hipStream_t stream) {
  (void)in_sizes; (void)n_in; (void)out_size; (void)ws_size;
  const float* x  = (const float*)d_in[0];
  const float* Wq = (const float*)d_in[1];
  const float* bq = (const float*)d_in[2];
  const float* Wk = (const float*)d_in[3];
  const float* bk = (const float*)d_in[4];
  float* out = (float*)d_out;

  const size_t nX = (size_t)TOKENS * DMODEL;    // 3,145,728
  const size_t nW = (size_t)DMODEL * DMODEL;    //   589,824

  __bf16* wsX  = (__bf16*)d_ws;
  __bf16* wsWq = wsX  + nX;
  __bf16* wsWk = wsWq + nW;
  __bf16* wsQ  = wsWk + nW;
  __bf16* wsK  = wsQ  + nX;

  cvt_bf16_kernel<<<dim3((unsigned)(nX / 8 / 256)), 256, 0, stream>>>(x,  wsX,  (int)nX);
  cvt_bf16_kernel<<<dim3((unsigned)(nW / 8 / 256)), 256, 0, stream>>>(Wq, wsWq, (int)nW);
  cvt_bf16_kernel<<<dim3((unsigned)(nW / 8 / 256)), 256, 0, stream>>>(Wk, wsWk, (int)nW);

  qk_proj_kernel<<<dim3(TOKENS / 16, 12), 256, 0, stream>>>(wsX, wsWq, bq, wsWk, bk,
                                                            wsQ, wsK);
  attn_kernel<<<dim3(SEQ / 16, 2), 256, 0, stream>>>(wsQ, wsK, out);
}
